// CouplingLayer_64209761075306
// MI455X (gfx1250) — matrix operand order repaired and checked by reading, not compiled, not run on hardware
//
#include <hip/hip_runtime.h>

// ---------------------------------------------------------------------------
// CDNA5 (gfx1250, wave32) implementation of the coupling-layer flow.
// GEMMs run on v_wmma_f32_16x16x32_bf16 (f32 accumulate); spline math in f32.
// Block tile 128x256, 8 waves, 64x64 per wave (16 WMMA : 16 ds_load_b128).
// Global->LDS staging uses GLOBAL_LOAD_ASYNC_TO_LDS_B128 when available.
// ---------------------------------------------------------------------------

typedef __attribute__((ext_vector_type(16))) __bf16 v16bf;
typedef __attribute__((ext_vector_type(8)))  float  v8f;
typedef __attribute__((ext_vector_type(4)))  int    v4i_vec;

typedef __attribute__((address_space(1))) v4i_vec* as1_v4i;
typedef __attribute__((address_space(3))) v4i_vec* as3_v4i;

#define B_ROWS   32768
#define D_IN     128
#define D_HALF   64
#define HID      2048
#define D_OUT3   1536   // 64 dims * 24 params

#define MBLK 128
#define NBLK 256
#define KBLK 32
#define LDT  40         // padded LDS row stride in ushorts (80B, conflict-free)

#if defined(__has_builtin)
#  if __has_builtin(__builtin_amdgcn_global_load_async_to_lds_b128) && \
      __has_builtin(__builtin_amdgcn_s_wait_asynccnt)
#    define USE_ASYNC_COPY 1
#  endif
#endif
#ifndef USE_ASYNC_COPY
#  define USE_ASYNC_COPY 0
#endif

static __device__ __forceinline__ unsigned short f32_to_bf16_rne(float f) {
    unsigned int u = __float_as_uint(f);
    unsigned int r = u + 0x7FFFu + ((u >> 16) & 1u);     // round-to-nearest-even
    if ((u & 0x7F800000u) == 0x7F800000u) r = u;         // inf/NaN passthrough
    return (unsigned short)(r >> 16);
}

// 16-byte global -> LDS copy; async (ASYNCcnt) path on gfx1250 toolchains that
// expose the builtin, register-staged fallback otherwise.
// Global pointer: addrspace(1), identical address value -> uintptr roundtrip.
// LDS pointer: addrspace(3) is the 32-bit LDS offset; generic LDS addresses
// carry that offset in addr[31:0] (ISA 10.2 aperture rules), so truncation is
// the correct conversion.
static __device__ __forceinline__ void copy16_g2l(const unsigned short* g, unsigned short* l) {
#if USE_ASYNC_COPY
    __builtin_amdgcn_global_load_async_to_lds_b128(
        (as1_v4i)(uintptr_t)g,
        (as3_v4i)(unsigned int)(uintptr_t)l,
        0, 0);
#else
    *(uint4*)l = *(const uint4*)g;
#endif
}

static __device__ __forceinline__ void copy_fence_then_barrier() {
#if USE_ASYNC_COPY
    __builtin_amdgcn_s_wait_asynccnt(0);
#endif
    __syncthreads();
}

// ---- pack transform half of x (even columns) into bf16 ---------------------
__global__ __launch_bounds__(256)
void pack_x_kernel(const float* __restrict__ x, unsigned short* __restrict__ Xt) {
    long i = (long)blockIdx.x * 256 + threadIdx.x;       // over B*64
    if (i >= (long)B_ROWS * D_HALF) return;
    long b = i / D_HALF, j = i % D_HALF;
    Xt[i] = f32_to_bf16_rne(x[b * D_IN + 2 * j]);
}

// ---- W (K x N, f32 row-major) -> Wt (N x K, bf16 row-major) ----------------
__global__ __launch_bounds__(256)
void transpose_bf16_kernel(const float* __restrict__ W, unsigned short* __restrict__ Wt,
                           int K, int N) {
    long i = (long)blockIdx.x * 256 + threadIdx.x;
    if (i >= (long)K * N) return;
    long k = i / N, n = i % N;
    Wt[n * (long)K + k] = f32_to_bf16_rne(W[i]);
}

// ---- tiled bf16 WMMA GEMM:  Out = epilogue(A[MxK] * Bt[NxK]^T) -------------
// mode 0: v = gamma*relu(v+bias)*inv + beta  -> bf16
// mode 1: v = v + bias                       -> f32
__global__ __launch_bounds__(256)
void gemm_bf16_wmma_kernel(const unsigned short* __restrict__ A,
                           const unsigned short* __restrict__ Bt,
                           void* __restrict__ Out,
                           const float* __restrict__ bias,
                           const float* __restrict__ gamma,
                           const float* __restrict__ beta,
                           int M, int N, int K, int mode) {
    __shared__ __align__(16) unsigned short As[MBLK * LDT];   // 10.0 KB
    __shared__ __align__(16) unsigned short Bs[NBLK * LDT];   // 20.0 KB

    const int tid   = threadIdx.x;
    const int lane  = tid & 31;
    const int wid   = tid >> 5;      // 0..7
    const int waveM = wid & 1;       // 0..1  -> 64 rows each
    const int waveN = wid >> 1;      // 0..3  -> 64 cols each
    const int lrow  = lane & 15;
    const int khalf = lane >> 4;

    const long blockM = (long)blockIdx.y * MBLK;
    const long blockN = (long)blockIdx.x * NBLK;

    v8f acc[4][4];
#pragma unroll
    for (int mt = 0; mt < 4; ++mt)
#pragma unroll
        for (int nt = 0; nt < 4; ++nt)
            acc[mt][nt] = (v8f){0.f, 0.f, 0.f, 0.f, 0.f, 0.f, 0.f, 0.f};

    union Frag { v16bf v; uint4 u[2]; };

    for (int k0 = 0; k0 < K; k0 += KBLK) {
        // stage A (128x32) and Bt-slice (256x32); 16B chunks, 6 per thread
#pragma unroll
        for (int p = 0; p < 2; ++p) {            // A: 512 chunks
            int c = tid + p * 256;
            int r = c >> 2;
            int col = (c & 3) * 8;
            copy16_g2l(A + (blockM + r) * (long)K + k0 + col, &As[r * LDT + col]);
        }
#pragma unroll
        for (int p = 0; p < 4; ++p) {            // B: 1024 chunks
            int c = tid + p * 256;
            int r = c >> 2;
            int col = (c & 3) * 8;
            copy16_g2l(Bt + (blockN + r) * (long)K + k0 + col, &Bs[r * LDT + col]);
        }
        copy_fence_then_barrier();

        // B fragments: lane half khalf holds K = khalf*16 + 2v + {0,1}
        Frag bf[4];
#pragma unroll
        for (int nt = 0; nt < 4; ++nt) {
            const unsigned short* bp = &Bs[(waveN * 64 + nt * 16 + lrow) * LDT + khalf * 16];
            bf[nt].u[0] = *(const uint4*)(bp);
            bf[nt].u[1] = *(const uint4*)(bp + 8);
        }
        // A fragments: lane half khalf holds K in [khalf*16, +16)
#pragma unroll
        for (int mt = 0; mt < 4; ++mt) {
            Frag af;
            const unsigned short* ap = &As[(waveM * 64 + mt * 16 + lrow) * LDT + khalf * 16];
            af.u[0] = *(const uint4*)(ap);
            af.u[1] = *(const uint4*)(ap + 8);
#pragma unroll
            for (int nt = 0; nt < 4; ++nt) {
                acc[mt][nt] = __builtin_amdgcn_wmma_f32_16x16x32_bf16(
                    false, af.v, false, bf[nt].v, (short)0, acc[mt][nt], false, false);
            }
        }
        __syncthreads();
    }

    const float inv = rsqrtf(1.0f + 1e-5f);
#pragma unroll
    for (int mt = 0; mt < 4; ++mt) {
#pragma unroll
        for (int nt = 0; nt < 4; ++nt) {
            long n = blockN + waveN * 64 + nt * 16 + lrow;
            float bsv = bias  ? bias[n]  : 0.0f;
            float gv  = gamma ? gamma[n] : 1.0f;
            float bev = beta  ? beta[n]  : 0.0f;
#pragma unroll
            for (int r = 0; r < 8; ++r) {
                long m = blockM + waveM * 64 + mt * 16 + khalf * 8 + r;
                float v = acc[mt][nt][r];
                if (mode == 0) {
                    v += bsv;
                    v = v > 0.0f ? v : 0.0f;
                    v = gv * v * inv + bev;
                    ((unsigned short*)Out)[m * (long)N + n] = f32_to_bf16_rne(v);
                } else {
                    ((float*)Out)[m * (long)N + n] = v + bsv;
                }
            }
        }
    }
}

// ---- rational-quadratic spline + output assembly (one wave per batch row) --
__global__ __launch_bounds__(256)
void spline_kernel(const float* __restrict__ x, const float* __restrict__ P,
                   float* __restrict__ out, float* __restrict__ logdet) {
    const int  lane = threadIdx.x & 31;
    const long row  = (long)blockIdx.x * 8 + (threadIdx.x >> 5);
    float lsum = 0.0f;
#pragma unroll
    for (int rep = 0; rep < 2; ++rep) {
        const int j = lane + rep * 32;                  // identity dim 0..63
        const float* p = P + row * (long)D_OUT3 + (long)j * 24;

        float ew[8], eh[8], d[9], cw[9], ch[9];
        float mw = -1e30f, mh = -1e30f;
#pragma unroll
        for (int i = 0; i < 8; ++i) { mw = fmaxf(mw, p[i]); mh = fmaxf(mh, p[8 + i]); }
        float sw = 0.f, sh = 0.f;
#pragma unroll
        for (int i = 0; i < 8; ++i) {
            ew[i] = __expf(p[i] - mw);      sw += ew[i];
            eh[i] = __expf(p[8 + i] - mh);  sh += eh[i];
        }
        const float rw = (1.0f - 0.001f * 8.0f) / sw;
        const float rh = (1.0f - 0.001f * 8.0f) / sh;
        cw[0] = 0.f; ch[0] = 0.f;
#pragma unroll
        for (int i = 0; i < 8; ++i) {
            cw[i + 1] = cw[i] + (0.001f + rw * ew[i]);
            ch[i + 1] = ch[i] + (0.001f + rh * eh[i]);
        }
#pragma unroll
        for (int i = 0; i < 8; ++i) {
            float v = p[16 + i];
            float sp = (v > 20.0f) ? v : log1pf(__expf(v));  // softplus
            d[i] = 0.001f + sp;
        }
        d[8] = 0.001f;                                   // pad value = MIN_D

        const float xi = x[row * D_IN + 2 * j + 1];
        out[row * D_IN + 2 * j] = x[row * D_IN + 2 * j]; // passthrough half

        int bin = 0;
#pragma unroll
        for (int i = 0; i < 8; ++i) bin += (xi > cw[i]) ? 1 : 0;  // sum(x > cw[:-1])
        bin = bin < 0 ? 0 : (bin > 7 ? 7 : bin);

        const float xl = cw[bin], xr = cw[bin + 1];
        const float yl = ch[bin], yr = ch[bin + 1];
        const float bw = xr - xl, bh = yr - yl;
        const float dl = d[bin],  dr = d[bin + 1];
        float t = (xi - xl) / bw;
        t = fminf(fmaxf(t, 0.0f), 1.0f);
        const float num = bh * (dl * t * t + 2.0f * t * (1.0f - t));
        const float den = dl + (dr - dl) * t;
        out[row * D_IN + 2 * j + 1] = yl + num / den;
        lsum += logf(bh) + 2.0f * logf(2.0f * t * (1.0f - t) * dr + dl) - logf(den);
    }
#pragma unroll
    for (int off = 16; off > 0; off >>= 1) lsum += __shfl_xor(lsum, off, 32);
    if (lane == 0) logdet[row] = lsum;
}

// ---------------------------------------------------------------------------
extern "C" void kernel_launch(void* const* d_in, const int* in_sizes, int n_in,
                              void* d_out, int out_size, void* d_ws, size_t ws_size,
                              hipStream_t stream) {
    const float* x  = (const float*)d_in[0];
    const float* W1 = (const float*)d_in[1];
    const float* b1 = (const float*)d_in[2];
    const float* g1 = (const float*)d_in[3];
    const float* be1= (const float*)d_in[4];
    const float* W2 = (const float*)d_in[5];
    const float* b2 = (const float*)d_in[6];
    const float* g2 = (const float*)d_in[7];
    const float* be2= (const float*)d_in[8];
    const float* W3 = (const float*)d_in[9];
    const float* b3 = (const float*)d_in[10];

    char* ws = (char*)d_ws;
    // workspace layout (all offsets 256B aligned)
    size_t off = 0;
    unsigned short* Xt  = (unsigned short*)(ws + off); off += (size_t)B_ROWS * D_HALF * 2;        //   4 MB
    unsigned short* W1t = (unsigned short*)(ws + off); off += (size_t)HID * D_HALF * 2;           // 256 KB
    unsigned short* W2t = (unsigned short*)(ws + off); off += (size_t)HID * HID * 2;              //   8 MB
    unsigned short* W3t = (unsigned short*)(ws + off); off += (size_t)D_OUT3 * HID * 2;           //   6 MB
    unsigned short* H1  = (unsigned short*)(ws + off); off += (size_t)B_ROWS * HID * 2;           // 128 MB
    unsigned short* H2  = (unsigned short*)(ws + off); off += (size_t)B_ROWS * HID * 2;           // 128 MB
    float*          Pp  = (float*)(ws + off);          off += (size_t)B_ROWS * D_OUT3 * 4;        // 192 MB
    (void)ws_size; (void)in_sizes; (void)n_in; (void)out_size;

    // 1) pack / transpose to bf16
    pack_x_kernel<<<(B_ROWS * D_HALF) / 256, 256, 0, stream>>>(x, Xt);
    transpose_bf16_kernel<<<(D_HALF * HID) / 256, 256, 0, stream>>>(W1, W1t, D_HALF, HID);
    transpose_bf16_kernel<<<(HID * HID) / 256,    256, 0, stream>>>(W2, W2t, HID, HID);
    transpose_bf16_kernel<<<(HID * D_OUT3) / 256, 256, 0, stream>>>(W3, W3t, HID, D_OUT3);

    // 2) MLP: three WMMA GEMMs
    dim3 blk(256);
    dim3 g1d(HID / NBLK,    B_ROWS / MBLK);   // (8, 256)
    dim3 g2d(HID / NBLK,    B_ROWS / MBLK);   // (8, 256)
    dim3 g3d(D_OUT3 / NBLK, B_ROWS / MBLK);   // (6, 256)
    gemm_bf16_wmma_kernel<<<g1d, blk, 0, stream>>>(Xt, W1t, (void*)H1, b1, g1, be1,
                                                   B_ROWS, HID, D_HALF, 0);
    gemm_bf16_wmma_kernel<<<g2d, blk, 0, stream>>>(H1, W2t, (void*)H2, b2, g2, be2,
                                                   B_ROWS, HID, HID, 0);
    gemm_bf16_wmma_kernel<<<g3d, blk, 0, stream>>>(H2, W3t, (void*)Pp, b3, nullptr, nullptr,
                                                   B_ROWS, D_OUT3, HID, 1);

    // 3) spline + output assembly; log_det appended after the (B,128) output
    float* out_main = (float*)d_out;
    float* out_ld   = out_main + (size_t)B_ROWS * D_IN;
    spline_kernel<<<B_ROWS / 8, 256, 0, stream>>>(x, Pp, out_main, out_ld);
}
